// HINAC_53704271069641
// MI455X (gfx1250) — matrix operand order repaired and checked by reading, not compiled, not run on hardware
//
#include <hip/hip_runtime.h>

#define NN     40000
#define NEDGE  320000
#define DD     256
#define BBATCH 1024
#define LSEQ   30
#define BL     (BBATCH*LSEQ)
#define NHEADS 4
#define HD     64

typedef __bf16 bf16;
typedef bf16  v16bf __attribute__((ext_vector_type(16)));
typedef bf16  v8bf  __attribute__((ext_vector_type(8)));
typedef float v8f   __attribute__((ext_vector_type(8)));
typedef float v4f   __attribute__((ext_vector_type(4)));

// Async copy of 16 bytes global -> LDS (CDNA5 GLOBAL_LOAD_ASYNC_TO_LDS_B128,
// tracked per-wave by ASYNCcnt). VDST operand = workgroup-relative LDS byte
// address (LDS aperture keeps the offset in addr[31:0] of a generic pointer).
__device__ __forceinline__ void async_copy_b128(const void* gptr, void* lptr)
{
    unsigned long long ga = (unsigned long long)(uintptr_t)gptr;
    unsigned int       la = (unsigned int)(uintptr_t)lptr;
    asm volatile("global_load_async_to_lds_b128 %0, %1, off"
                 :: "v"(la), "v"(ga) : "memory");
}
__device__ __forceinline__ void async_wait0()
{
    asm volatile("s_wait_asynccnt 0x0" ::: "memory");
}

// ---------------------------------------------------------------------------
// Weight pre-pass: W[K][N] fp32  ->  Wt[N][K] bf16   (once per launch)
// ---------------------------------------------------------------------------
__global__ void wconv_kernel(const float* __restrict__ W, bf16* __restrict__ Wt,
                             int K, int N)
{
    int i = blockIdx.x * blockDim.x + threadIdx.x;
    if (i < K * N) {
        int k = i / N, n = i % N;
        Wt[(size_t)n * K + k] = (bf16)W[(size_t)k * N + n];
    }
}

// ---------------------------------------------------------------------------
// bf16-WMMA GEMM:  C[M,N] = epi( A[M,K] (*rsA row) @ B[K,N] )
//   A : fp32 activations, async double-buffered into LDS, converted per-frag
//   Bt: pre-converted bf16 [N][K], async double-buffered (contiguous rows)
// Epilogue flags are compile-time. 256 threads = 8 waves; wave tile 16x32
// (2 WMMAs), block tile 32x128, K step 32, LDS ping-pong prefetch.
// ---------------------------------------------------------------------------
template<bool RSA, bool RSO, bool BIAS, bool RES, bool RELU>
__global__ __launch_bounds__(256)
void gemm_bf16_wmma(const float* __restrict__ A, const bf16* __restrict__ Bt,
                    float* __restrict__ C, int M, int N, int K,
                    const float* __restrict__ rsA, const float* __restrict__ rsOut,
                    const float* __restrict__ bias, const float* __restrict__ res)
{
    __shared__ alignas(16) float As[2][32][32];    // fp32 [m][k]  2 x 4 KB
    __shared__ alignas(16) bf16  Bs[2][128][32];   // bf16 [n][k]  2 x 8 KB

    const int tid  = threadIdx.x;
    const int lane = tid & 31;
    const int w    = tid >> 5;          // wave 0..7 (wave32)
    const int wm   = w >> 2;            // 0..1 : 16-row tile
    const int wn   = w & 3;             // 0..3 : 32-col tile
    const int m0   = blockIdx.y * 32;
    const int n0   = blockIdx.x * 128;
    const int lrow = lane & 15;
    const int hi   = lane >> 4;

    // staging assignments (fixed per thread)
    const int amm = tid >> 3;           // A row 0..31
    const int akk = (tid & 7) * 4;      // A col chunk (4 floats = 16B)
    const int bnn = tid >> 1;           // B row (n) 0..127
    const int bk0 = (tid & 1) * 16;     // B col chunk base (bf16 elems)
    int agm = m0 + amm; if (agm > M - 1) agm = M - 1;   // row clamp (OOB rows discarded)
    const float* aRowG = A + (size_t)agm * K;
    const bf16*  bRowG = Bt + (size_t)(n0 + bnn) * K;

    // per-lane fragment row + its scale, hoisted out of the K loop
    int frow = m0 + wm * 16 + lrow; if (frow > M - 1) frow = M - 1;
    float rs = 1.0f;
    if constexpr (RSA) rs = rsA[frow];

    v8f acc0 = {0.f,0.f,0.f,0.f,0.f,0.f,0.f,0.f};
    v8f acc1 = {0.f,0.f,0.f,0.f,0.f,0.f,0.f,0.f};

    auto stage = [&](int buf, int k0) {
        async_copy_b128(aRowG + k0 + akk,     &As[buf][amm][akk]);
        async_copy_b128(bRowG + k0 + bk0,     &Bs[buf][bnn][bk0]);
        async_copy_b128(bRowG + k0 + bk0 + 8, &Bs[buf][bnn][bk0 + 8]);
    };

    stage(0, 0);
    int cur = 0;
    for (int k0 = 0; k0 < K; k0 += 32) {
        async_wait0();        // own copies done ...
        __syncthreads();      // ... barrier => all waves' copies visible
        if (k0 + 32 < K) stage(cur ^ 1, k0 + 32);   // prefetch next tile

        // A fragment: row m = wm*16+lrow ; K = 16*hi..16*hi+15
        const float* arow = &As[cur][wm * 16 + lrow][0];
        v4f a0 = *(const v4f*)(arow + 16 * hi);
        v4f a1 = *(const v4f*)(arow + 16 * hi + 4);
        v4f a2 = *(const v4f*)(arow + 16 * hi + 8);
        v4f a3 = *(const v4f*)(arow + 16 * hi + 12);
        v16bf a;
        a[0]  = (bf16)(a0[0] * rs); a[1]  = (bf16)(a0[1] * rs);
        a[2]  = (bf16)(a0[2] * rs); a[3]  = (bf16)(a0[3] * rs);
        a[4]  = (bf16)(a1[0] * rs); a[5]  = (bf16)(a1[1] * rs);
        a[6]  = (bf16)(a1[2] * rs); a[7]  = (bf16)(a1[3] * rs);
        a[8]  = (bf16)(a2[0] * rs); a[9]  = (bf16)(a2[1] * rs);
        a[10] = (bf16)(a2[2] * rs); a[11] = (bf16)(a2[3] * rs);
        a[12] = (bf16)(a3[0] * rs); a[13] = (bf16)(a3[1] * rs);
        a[14] = (bf16)(a3[2] * rs); a[15] = (bf16)(a3[3] * rs);

        // B fragments: col n = wn*32 + f*16 + lrow ; K = 16*hi..16*hi+15
        const bf16* brow0 = &Bs[cur][wn * 32 + lrow][0] + 16 * hi;
        v8bf b00 = *(const v8bf*)brow0;
        v8bf b01 = *(const v8bf*)(brow0 + 8);
        v16bf b0 = __builtin_shufflevector(b00, b01, 0,1,2,3,4,5,6,7,8,9,10,11,12,13,14,15);
        const bf16* brow1 = &Bs[cur][wn * 32 + 16 + lrow][0] + 16 * hi;
        v8bf b10 = *(const v8bf*)brow1;
        v8bf b11 = *(const v8bf*)(brow1 + 8);
        v16bf b1 = __builtin_shufflevector(b10, b11, 0,1,2,3,4,5,6,7,8,9,10,11,12,13,14,15);

        acc0 = __builtin_amdgcn_wmma_f32_16x16x32_bf16(false, a, false, b0, (short)0, acc0, false, false);
        acc1 = __builtin_amdgcn_wmma_f32_16x16x32_bf16(false, a, false, b1, (short)0, acc1, false, false);
        cur ^= 1;
    }

    // C/D layout: VGPR g -> row = g + 8*hi, col = lane&15
    for (int f = 0; f < 2; ++f) {
        const v8f accv = f ? acc1 : acc0;
        const int gn = n0 + wn * 32 + f * 16 + lrow;
        if (gn < N) {
            for (int g = 0; g < 8; ++g) {
                int gm = m0 + wm * 16 + g + 8 * hi;
                if (gm < M) {
                    float v = accv[g];
                    if constexpr (RSO)  v *= rsOut[gm];
                    if constexpr (BIAS) v += bias[gn];
                    if constexpr (RES)  v += res[(size_t)gm * N + gn];
                    if constexpr (RELU) v = fmaxf(v, 0.f);
                    C[(size_t)gm * N + gn] = v;
                }
            }
        }
    }
}

// ---------------------------------------------------------------------------
// Degree + norm
// ---------------------------------------------------------------------------
__global__ void deg_kernel(const int* __restrict__ src, const int* __restrict__ dst,
                           float* __restrict__ odeg, float* __restrict__ ideg, int E)
{
    int i = blockIdx.x * blockDim.x + threadIdx.x;
    if (i < E) {
        atomicAdd(&odeg[src[i]], 1.0f);
        atomicAdd(&ideg[dst[i]], 1.0f);
    }
}

__global__ void norm_kernel(const float* __restrict__ odeg, const float* __restrict__ ideg,
                            float* __restrict__ no, float* __restrict__ ni,
                            float* __restrict__ invin, int n)
{
    int v = blockIdx.x * blockDim.x + threadIdx.x;
    if (v < n) {
        float od  = fmaxf(odeg[v], 1.0f);
        float idg = fmaxf(ideg[v], 1.0f);
        no[v]    = rsqrtf(od);
        ni[v]    = rsqrtf(idg);
        invin[v] = 1.0f / idg;
    }
}

// ---------------------------------------------------------------------------
// HADE: pp = softmax(relu(type_emb[t] @ mlpW + mlpb) @ protoW + protob)
// ---------------------------------------------------------------------------
__global__ __launch_bounds__(256)
void hade_pp_kernel(const int* __restrict__ node_type, const float* __restrict__ type_emb,
                    const float* __restrict__ mlpW, const float* __restrict__ mlpb,
                    const float* __restrict__ protoW, const float* __restrict__ protob,
                    float* __restrict__ pp)
{
    __shared__ float sz[256];
    __shared__ float slog[4];
    int v = blockIdx.x;
    int tid = threadIdx.x;
    int t = node_type[v];
    float r0c0 = type_emb[t*4+0], r0c1 = type_emb[t*4+1];
    float r0c2 = type_emb[t*4+2], r0c3 = type_emb[t*4+3];
    float z = mlpb[tid];
    z += r0c0 * mlpW[0*DD + tid];
    z += r0c1 * mlpW[1*DD + tid];
    z += r0c2 * mlpW[2*DD + tid];
    z += r0c3 * mlpW[3*DD + tid];
    sz[tid] = fmaxf(z, 0.f);
    __syncthreads();
    if (tid < 4) {
        float acc = protob[tid];
        for (int d = 0; d < DD; ++d) acc += sz[d] * protoW[d*4 + tid];
        slog[tid] = acc;
    }
    __syncthreads();
    if (tid == 0) {
        float m = fmaxf(fmaxf(slog[0], slog[1]), fmaxf(slog[2], slog[3]));
        float e0 = expf(slog[0]-m), e1 = expf(slog[1]-m);
        float e2 = expf(slog[2]-m), e3 = expf(slog[3]-m);
        float inv = 1.0f / (e0+e1+e2+e3);
        pp[(size_t)v*4+0] = e0*inv; pp[(size_t)v*4+1] = e1*inv;
        pp[(size_t)v*4+2] = e2*inv; pp[(size_t)v*4+3] = e3*inv;
    }
}

__global__ void pp_scatter_kernel(const float* __restrict__ pp, const int* __restrict__ src,
                                  const int* __restrict__ dst, float* __restrict__ nd, int E)
{
    int i = blockIdx.x * blockDim.x + threadIdx.x;
    if (i < E*4) {
        int e = i >> 2, c = i & 3;
        atomicAdd(&nd[(size_t)dst[e]*4 + c], pp[(size_t)src[e]*4 + c]);
    }
}

__global__ void hade_final_kernel(const float* __restrict__ nd, const float* __restrict__ invin,
                                  const float* __restrict__ fW, const float* __restrict__ fb,
                                  float* __restrict__ r, int n)
{
    int i = blockIdx.x * blockDim.x + threadIdx.x;
    if (i < n) {
        int v = i >> 8, d = i & 255;
        float inv = invin[v];
        float acc = fb[d];
        acc += nd[(size_t)v*4+0]*inv * fW[0*DD + d];
        acc += nd[(size_t)v*4+1]*inv * fW[1*DD + d];
        acc += nd[(size_t)v*4+2]*inv * fW[2*DD + d];
        acc += nd[(size_t)v*4+3]*inv * fW[3*DD + d];
        r[i] = fmaxf(acc, 0.f);
    }
}

// ---------------------------------------------------------------------------
// Edge aggregation: out[dst] += x[src] * (rowscale?rowscale[src]:1)  (256 ch)
// ---------------------------------------------------------------------------
__global__ __launch_bounds__(256)
void scatter256_kernel(const float* __restrict__ x, const float* __restrict__ rowscale,
                       const int* __restrict__ src, const int* __restrict__ dst,
                       float* __restrict__ out)
{
    int e = blockIdx.x;
    int d = threadIdx.x;
    int s = src[e], t = dst[e];
    float v = x[(size_t)s*DD + d];
    if (rowscale) v *= rowscale[s];
    atomicAdd(&out[(size_t)t*DD + d], v);
}

__global__ void wtl_kernel(const int* __restrict__ node_type, const float* __restrict__ wt4,
                           float* __restrict__ wtl, int n)
{
    int v = blockIdx.x * blockDim.x + threadIdx.x;
    if (v < n) wtl[v] = wt4[node_type[v]];
}

__global__ void rpost_kernel(const float* __restrict__ tmp2, const float* __restrict__ ni,
                             const float* __restrict__ reb, float* __restrict__ r, int n)
{
    int i = blockIdx.x * blockDim.x + threadIdx.x;
    if (i < n) {
        int v = i >> 8, d = i & 255;
        r[i] = fmaxf(tmp2[i] * ni[v] + reb[d], 0.f);
    }
}

__global__ void gather_kernel(const float* __restrict__ gh, const float* __restrict__ r,
                              const int* __restrict__ seqs, float* __restrict__ h,
                              float* __restrict__ rh, int n)
{
    int i = blockIdx.x * blockDim.x + threadIdx.x;
    if (i < n) {
        int s = i >> 8, d = i & 255;
        int node = seqs[s];
        h[i]  = gh[(size_t)node*DD + d];
        rh[i] = r[(size_t)node*DD + d];
    }
}

// ---------------------------------------------------------------------------
// Attention block per (batch, head)
// ---------------------------------------------------------------------------
__global__ __launch_bounds__(256)
void attn_kernel(const float* __restrict__ fl, const float* __restrict__ fr,
                 const float* __restrict__ rk, const float* __restrict__ rt,
                 const float* __restrict__ al, const float* __restrict__ ar,
                 float* __restrict__ ctx)
{
    __shared__ float s_rk[LSEQ][DD];     // 30 KB
    __shared__ float s_fr[LSEQ][HD];     // 7.5 KB
    __shared__ float s_sc[LSEQ][LSEQ];   // 3.6 KB
    __shared__ float s_sl[LSEQ], s_sr[LSEQ];

    int bh = blockIdx.x;
    int b  = bh >> 2;
    int hh = bh & 3;
    int tid = threadIdx.x;
    size_t base = (size_t)b * LSEQ;

    for (int i = tid; i < LSEQ*DD; i += 256) {
        int li = i >> 8, d = i & 255;
        s_rk[li][d] = rk[(base + li)*1024 + hh*DD + d];
    }
    for (int i = tid; i < LSEQ*HD; i += 256) {
        int li = i >> 6, d = i & 63;
        s_fr[li][d] = fr[(base + li)*DD + hh*HD + d];
    }
    __syncthreads();

    if (tid < LSEQ) {
        float sl = 0.f, sr = 0.f;
        for (int d = 0; d < HD; ++d) {
            float xl = fl[(base + tid)*DD + hh*HD + d];
            xl = xl > 0.f ? xl : 0.01f * xl;
            sl += xl * al[d];
            float xr = s_fr[tid][d];
            xr = xr > 0.f ? xr : 0.01f * xr;
            sr += xr * ar[d];
        }
        s_sl[tid] = sl; s_sr[tid] = sr;
    }
    __syncthreads();

    for (int p = tid; p < LSEQ*LSEQ; p += 256) {
        int i = p / LSEQ, j = p % LSEQ;
        const float* rtp = &rt[(base + j)*1024 + hh*DD];
        float acc = 0.f;
        for (int d = 0; d < DD; ++d) acc += s_rk[i][d] * rtp[d];
        s_sc[i][j] = acc + s_sl[i] + s_sr[j];
    }
    __syncthreads();

    if (tid < LSEQ) {
        float m = -1e30f;
        for (int j = 0; j < LSEQ; ++j) m = fmaxf(m, s_sc[tid][j]);
        float s = 0.f;
        for (int j = 0; j < LSEQ; ++j) { float e = expf(s_sc[tid][j] - m); s_sc[tid][j] = e; s += e; }
        float inv = 1.0f / s;
        for (int j = 0; j < LSEQ; ++j) s_sc[tid][j] *= inv;
    }
    __syncthreads();

    for (int p = tid; p < LSEQ*HD; p += 256) {
        int i = p >> 6, d = p & 63;
        float acc = 0.f;
        for (int j = 0; j < LSEQ; ++j) acc += s_sc[i][j] * s_fr[j][d];
        ctx[(base + i)*DD + hh*HD + d] = acc;
    }
}

// ---------------------------------------------------------------------------
// LayerNorm over 256 features; one block per row
// ---------------------------------------------------------------------------
__global__ __launch_bounds__(256)
void ln_kernel(const float* __restrict__ x, const float* __restrict__ g,
               const float* __restrict__ b, float* __restrict__ out)
{
    __shared__ float sred[256];
    int row = blockIdx.x;
    int tid = threadIdx.x;
    float v = x[(size_t)row*DD + tid];
    sred[tid] = v;
    __syncthreads();
    for (int s = 128; s > 0; s >>= 1) { if (tid < s) sred[tid] += sred[tid + s]; __syncthreads(); }
    float mu = sred[0] * (1.0f/DD);
    __syncthreads();
    float dv = v - mu;
    sred[tid] = dv * dv;
    __syncthreads();
    for (int s = 128; s > 0; s >>= 1) { if (tid < s) sred[tid] += sred[tid + s]; __syncthreads(); }
    float var = sred[0] * (1.0f/DD);
    out[(size_t)row*DD + tid] = dv * rsqrtf(var + 1e-5f) * g[tid] + b[tid];
}

__global__ void pred_kernel(const float* __restrict__ h, const float* __restrict__ pW,
                            const float* __restrict__ pb, float* __restrict__ out, int n)
{
    int i = blockIdx.x * blockDim.x + threadIdx.x;
    if (i < n) {
        int bb = i / 5, c = i % 5;
        const float* hr = &h[(size_t)bb * LSEQ * DD];   // row (b, l=0)
        float acc = pb[c];
        for (int d = 0; d < DD; ++d) acc += hr[d] * pW[d*5 + c];
        out[i] = acc;
    }
}

// ---------------------------------------------------------------------------
// Host
// ---------------------------------------------------------------------------
static inline void gemm(const float* A, const bf16* Bt, float* C, int M, int N, int K,
                        const float* rsA, const float* rsOut, const float* bias,
                        const float* res, int relu, hipStream_t s)
{
    dim3 g(N / 128, (M + 31) / 32);
    if (rsA && rsOut)                    // re: ft = (r*no)@W * wtl
        gemm_bf16_wmma<true,true,false,false,false><<<g,256,0,s>>>(A,Bt,C,M,N,K,rsA,rsOut,bias,res);
    else if (rsOut && bias && relu)      // gcn: relu(T@W * ni + b)
        gemm_bf16_wmma<false,true,true,false,true><<<g,256,0,s>>>(A,Bt,C,M,N,K,rsA,rsOut,bias,res);
    else if (bias)                       // fc: A@W + b
        gemm_bf16_wmma<false,false,true,false,false><<<g,256,0,s>>>(A,Bt,C,M,N,K,rsA,rsOut,bias,res);
    else if (res)                        // gt final: ctx@Wf + h
        gemm_bf16_wmma<false,false,false,true,false><<<g,256,0,s>>>(A,Bt,C,M,N,K,rsA,rsOut,bias,res);
    else                                 // plain
        gemm_bf16_wmma<false,false,false,false,false><<<g,256,0,s>>>(A,Bt,C,M,N,K,rsA,rsOut,bias,res);
}

extern "C" void kernel_launch(void* const* d_in, const int* in_sizes, int n_in,
                              void* d_out, int out_size, void* d_ws, size_t ws_size,
                              hipStream_t stream)
{
    const float* feat[4] = {(const float*)d_in[0], (const float*)d_in[1],
                            (const float*)d_in[2], (const float*)d_in[3]};
    const float* type_emb   = (const float*)d_in[4];
    const float* fc_W       = (const float*)d_in[5];
    const float* fc_b       = (const float*)d_in[6];
    const float* mlpW       = (const float*)d_in[7];
    const float* mlpb       = (const float*)d_in[8];
    const float* protoW     = (const float*)d_in[9];
    const float* protob     = (const float*)d_in[10];
    const float* finalW     = (const float*)d_in[11];
    const float* finalb     = (const float*)d_in[12];
    const float* gcn_W      = (const float*)d_in[13];
    const float* gcn_b      = (const float*)d_in[14];
    const float* re_W       = (const float*)d_in[15];
    const float* re_b       = (const float*)d_in[16];
    const float* re_wt      = (const float*)d_in[17];
    const float* gt_Wl      = (const float*)d_in[18];
    const float* gt_Wr      = (const float*)d_in[19];
    const float* gt_al      = (const float*)d_in[20];
    const float* gt_ar      = (const float*)d_in[21];
    const float* gt_rs      = (const float*)d_in[22];
    const float* gt_rt      = (const float*)d_in[23];
    const float* gt_Wf      = (const float*)d_in[24];
    const float* gt_ln_g    = (const float*)d_in[25];
    const float* gt_ln_b    = (const float*)d_in[26];
    const float* pred_W     = (const float*)d_in[27];
    const float* pred_b     = (const float*)d_in[28];
    const int*   seqs       = (const int*)d_in[29];
    const int*   node_type  = (const int*)d_in[30];
    const int*   src        = (const int*)d_in[31];
    const int*   dst        = (const int*)d_in[32];
    float*       out        = (float*)d_out;

    // workspace carve-out, 256B aligned
    char* base = (char*)d_ws;
    size_t off = 0;
    auto fall = [&](size_t nfloats) -> float* {
        float* p = (float*)(base + off);
        off += ((nfloats * sizeof(float)) + 255) & ~(size_t)255;
        return p;
    };
    auto hall = [&](size_t nbf) -> bf16* {
        bf16* p = (bf16*)(base + off);
        off += ((nbf * sizeof(bf16)) + 255) & ~(size_t)255;
        return p;
    };
    float* gh    = fall((size_t)NN * DD);   // node features; reused as x after gather
    float* rbuf  = fall((size_t)NN * DD);   // r
    float* tmp   = fall((size_t)NN * DD);   // agg scratch; reused as ctx
    float* ftb   = fall((size_t)NN * DD);   // ft; reused as fl
    float* tmp2  = fall((size_t)NN * DD);   // agg scratch 2; reused as fr
    float* hbuf  = fall((size_t)BL * DD);
    float* rhb   = fall((size_t)BL * DD);
    float* rkb   = fall((size_t)BL * 1024);
    float* rtb   = fall((size_t)BL * 1024);
    float* odeg  = fall(NN);
    float* ideg  = fall(NN);
    float* no    = fall(NN);
    float* ni    = fall(NN);
    float* invin = fall(NN);
    float* pp    = fall((size_t)NN * 4);
    float* nd    = fall((size_t)NN * 4);
    float* wtl   = fall(NN);

    // ---- pre-transpose+convert all GEMM weights to bf16 [N][K] ----
    auto wt = [&](const float* W, int K, int N) -> bf16* {
        bf16* p = hall((size_t)K * N);
        wconv_kernel<<<((size_t)K * N + 255) / 256, 256, 0, stream>>>(W, p, K, N);
        return p;
    };
    bf16* fcWt[4];
    for (int i = 0; i < 4; ++i) fcWt[i] = wt(fc_W + (size_t)i * 128 * DD, 128, DD);
    bf16 *gcnWt[4], *reWt[4];
    for (int l = 0; l < 4; ++l) {
        gcnWt[l] = wt(gcn_W + (size_t)l * DD * DD, DD, DD);
        reWt[l]  = wt(re_W  + (size_t)l * DD * DD, DD, DD);
    }
    bf16 *WlT[2], *WrT[2], *WfT[2], *rsT[2], *rtT[2];
    for (int l = 0; l < 2; ++l) {
        WlT[l] = wt(gt_Wl + (size_t)l * DD * DD, DD, DD);
        WrT[l] = wt(gt_Wr + (size_t)l * DD * DD, DD, DD);
        WfT[l] = wt(gt_Wf + (size_t)l * DD * DD, DD, DD);
        rsT[l] = wt(gt_rs + (size_t)l * DD * 1024, DD, 1024);
        rtT[l] = wt(gt_rt + (size_t)l * DD * 1024, DD, 1024);
    }

    // ---- degrees + norms ----
    hipMemsetAsync(odeg, 0, NN * sizeof(float), stream);
    hipMemsetAsync(ideg, 0, NN * sizeof(float), stream);
    hipMemsetAsync(nd,   0, (size_t)NN * 4 * sizeof(float), stream);
    deg_kernel<<<(NEDGE + 255) / 256, 256, 0, stream>>>(src, dst, odeg, ideg, NEDGE);
    norm_kernel<<<(NN + 255) / 256, 256, 0, stream>>>(odeg, ideg, no, ni, invin, NN);

    // ---- per-type input projections: gh ----
    for (int i = 0; i < 4; ++i)
        gemm(feat[i], fcWt[i], gh + (size_t)i * 10000 * DD,
             10000, DD, 128, nullptr, nullptr, fc_b + i * DD, nullptr, 0, stream);

    // ---- HADE ----
    hade_pp_kernel<<<NN, 256, 0, stream>>>(node_type, type_emb, mlpW, mlpb, protoW, protob, pp);
    pp_scatter_kernel<<<(NEDGE * 4 + 255) / 256, 256, 0, stream>>>(pp, src, dst, nd, NEDGE);
    hade_final_kernel<<<((size_t)NN * DD + 255) / 256, 256, 0, stream>>>(nd, invin, finalW, finalb, rbuf, NN * DD);

    // ---- GNN layers ----
    for (int l = 0; l < 4; ++l) {
        hipMemsetAsync(tmp, 0, (size_t)NN * DD * sizeof(float), stream);
        scatter256_kernel<<<NEDGE, 256, 0, stream>>>(gh, no, src, dst, tmp);
        gemm(tmp, gcnWt[l], gh, NN, DD, DD,
             nullptr, ni, gcn_b + l * DD, nullptr, 1, stream);
        wtl_kernel<<<(NN + 255) / 256, 256, 0, stream>>>(node_type, re_wt + l * 4, wtl, NN);
        gemm(rbuf, reWt[l], ftb, NN, DD, DD,
             no, wtl, nullptr, nullptr, 0, stream);
        hipMemsetAsync(tmp2, 0, (size_t)NN * DD * sizeof(float), stream);
        scatter256_kernel<<<NEDGE, 256, 0, stream>>>(ftb, nullptr, src, dst, tmp2);
        rpost_kernel<<<((size_t)NN * DD + 255) / 256, 256, 0, stream>>>(tmp2, ni, re_b + l * DD, rbuf, NN * DD);
    }

    // ---- gather sequences ----
    gather_kernel<<<((size_t)BL * DD + 255) / 256, 256, 0, stream>>>(gh, rbuf, seqs, hbuf, rhb, BL * DD);

    // ---- graph transformer layers ----
    float* flb  = ftb;   // reuse
    float* frb  = tmp2;  // reuse
    float* ctxb = tmp;   // reuse
    float* xb   = gh;    // reuse
    for (int l = 0; l < 2; ++l) {
        gemm(hbuf, WlT[l], flb, BL, DD, DD, nullptr, nullptr, nullptr, nullptr, 0, stream);
        gemm(hbuf, WrT[l], frb, BL, DD, DD, nullptr, nullptr, nullptr, nullptr, 0, stream);
        gemm(rhb,  rsT[l], rkb, BL, 1024, DD, nullptr, nullptr, nullptr, nullptr, 0, stream);
        gemm(rhb,  rtT[l], rtb, BL, 1024, DD, nullptr, nullptr, nullptr, nullptr, 0, stream);
        attn_kernel<<<BBATCH * NHEADS, 256, 0, stream>>>(flb, frb, rkb, rtb,
                                                         gt_al + l * HD, gt_ar + l * HD, ctxb);
        gemm(ctxb, WfT[l], xb, BL, DD, DD,
             nullptr, nullptr, nullptr, hbuf, 0, stream);
        ln_kernel<<<BL, 256, 0, stream>>>(xb, gt_ln_g + l * DD, gt_ln_b + l * DD, hbuf);
    }

    // ---- prediction head ----
    pred_kernel<<<(BBATCH * 5 + 255) / 256, 256, 0, stream>>>(hbuf, pred_W, pred_b, out, BBATCH * 5);
}
